// HVAE_65163243815707
// MI455X (gfx1250) — compile-verified, matrix operand order checked by reading, not run on hardware
//
#include <hip/hip_runtime.h>
#include <cstdint>

typedef unsigned short ushort_t;
typedef unsigned int uint_t;
typedef __attribute__((ext_vector_type(16))) __bf16 v16bf;
typedef __attribute__((ext_vector_type(8)))  float  v8f;
typedef __attribute__((ext_vector_type(4)))  uint_t v4u;
typedef __attribute__((ext_vector_type(4)))  int    v4i;

#define C_LOG2PI 1.8378770664093453f

#if defined(__has_builtin)
#if __has_builtin(__builtin_amdgcn_global_load_async_to_lds_b128) && \
    __has_builtin(__builtin_amdgcn_s_wait_asynccnt)
#define HAVE_ASYNC_LDS 1
#endif
#endif

#ifdef HAVE_ASYNC_LDS
// Builtin signature (from hipcc diagnostic): (v4i AS1*, v4i AS3*, imm offset, imm cpol)
typedef __attribute__((address_space(1))) v4i* as1_v4i_p;
typedef __attribute__((address_space(3))) v4i* as3_v4i_p;
// global (AS1) -> LDS (AS3) 16-byte async copy; LDS offset = low 32 bits of the
// generic LDS address (aperture rule), AS1 value == flat global address.
#define ASYNC_CP16(g, l)                                                  \
    __builtin_amdgcn_global_load_async_to_lds_b128(                       \
        (as1_v4i_p)(uintptr_t)(const void*)(g),                           \
        (as3_v4i_p)(uint32_t)(uintptr_t)(void*)(l),                       \
        0, 0)
#endif

__device__ __forceinline__ ushort_t f2bf(float f) {
    uint_t u = __float_as_uint(f);
    u += 0x7FFFu + ((u >> 16) & 1u);       // round-to-nearest-even
    return (ushort_t)(u >> 16);
}

union Frag { v4u q[2]; v16bf bf; };

// ---------------------------------------------------------------------------
// Tiled bf16 WMMA GEMM: C[M,N] = A[M,K] @ W[N,K]^T + bias[N]
// MODE 0: store f32 C.  MODE 1: fused recon loss vs xin, atomicAdd into acc.
// BM=128 BN=128 BK=32, 256 threads (8 waves), each wave owns a 32x64 C region.
// Double-buffered LDS; staging via async global->LDS when available.
// Out-of-range W rows are clamped: garbage only reaches C columns >= N,
// which the epilogue discards.
// ---------------------------------------------------------------------------
template<int MODE>
__global__ __launch_bounds__(256)
void gemm_bf16_kernel(const ushort_t* __restrict__ A, const ushort_t* __restrict__ W,
                      const float* __restrict__ bias, float* __restrict__ C,
                      const float* __restrict__ xin, float* __restrict__ acc,
                      int M, int N, int K) {
    __shared__ __align__(16) ushort_t ldsA[2 * 128 * 40];
    __shared__ __align__(16) ushort_t ldsB[2 * 128 * 40];
    __shared__ float red[256];

    const int tid  = threadIdx.x;
    const int lane = tid & 31;
    const int wave = tid >> 5;
    const int wm   = wave & 3;   // 4 waves along M (32 rows each)
    const int wn   = wave >> 2;  // 2 waves along N (64 cols each)
    const int m0   = blockIdx.y * 128;
    const int n0   = blockIdx.x * 128;

    v8f accv[2][4] = {};

    const int kb   = (lane >> 4) << 3;   // 0 or 8 (K offset per half-wave)
    const int rsel = lane & 15;

    // per-thread staging slots (straight-line, no loop)
    const int r0 = tid >> 2;             // 0..63
    const int c0 = (tid & 3) << 3;       // 0,8,16,24
    int nB0 = n0 + r0;       if (nB0 >= N) nB0 = N - 1;
    int nB1 = n0 + r0 + 64;  if (nB1 >= N) nB1 = N - 1;

    auto stage = [&](int buf, int k0) {
        ushort_t* lA = ldsA + buf * (128 * 40);
        ushort_t* lB = ldsB + buf * (128 * 40);
        const ushort_t* ga0 = A + ((size_t)(m0 + r0)      * K + (size_t)k0 + c0);
        const ushort_t* ga1 = A + ((size_t)(m0 + r0 + 64) * K + (size_t)k0 + c0);
        const ushort_t* gb0 = W + ((size_t)nB0 * K + (size_t)k0 + c0);
        const ushort_t* gb1 = W + ((size_t)nB1 * K + (size_t)k0 + c0);
        ushort_t* la0 = lA + r0 * 40 + c0;
        ushort_t* la1 = lA + (r0 + 64) * 40 + c0;
        ushort_t* lb0 = lB + r0 * 40 + c0;
        ushort_t* lb1 = lB + (r0 + 64) * 40 + c0;
#ifdef HAVE_ASYNC_LDS
        ASYNC_CP16(ga0, la0);
        ASYNC_CP16(ga1, la1);
        ASYNC_CP16(gb0, lb0);
        ASYNC_CP16(gb1, lb1);
#else
        v4u a0 = *(const v4u*)ga0;
        v4u a1 = *(const v4u*)ga1;
        v4u b0 = *(const v4u*)gb0;
        v4u b1 = *(const v4u*)gb1;
        *(v4u*)la0 = a0; *(v4u*)la1 = a1;
        *(v4u*)lb0 = b0; *(v4u*)lb1 = b1;
#endif
    };

    stage(0, 0);
#ifdef HAVE_ASYNC_LDS
    __builtin_amdgcn_s_wait_asynccnt(0);
#endif
    __syncthreads();

    int buf = 0;
    for (int k0 = 0; k0 < K; k0 += 32) {
        const bool has_next = (k0 + 32) < K;
        const ushort_t* lA = ldsA + buf * (128 * 40);
        const ushort_t* lB = ldsB + buf * (128 * 40);

        // ---- fragment loads first (DS in-order: wmma only waits on these) ----
        Frag af[2], bfr[4];
#pragma unroll
        for (int mt = 0; mt < 2; ++mt) {
            int row = (wm << 5) + (mt << 4) + rsel;
            af[mt].q[0] = *(const v4u*)&lA[row * 40 + kb];
            af[mt].q[1] = *(const v4u*)&lA[row * 40 + kb + 16];
        }
#pragma unroll
        for (int nt = 0; nt < 4; ++nt) {
            int col = (wn << 6) + (nt << 4) + rsel;
            bfr[nt].q[0] = *(const v4u*)&lB[col * 40 + kb];
            bfr[nt].q[1] = *(const v4u*)&lB[col * 40 + kb + 16];
        }

        // ---- stage next K-tile into the other buffer while we compute ----
        if (has_next) stage(buf ^ 1, k0 + 32);

#pragma unroll
        for (int mt = 0; mt < 2; ++mt)
#pragma unroll
            for (int nt = 0; nt < 4; ++nt)
                accv[mt][nt] = __builtin_amdgcn_wmma_f32_16x16x32_bf16(
                    false, af[mt].bf, false, bfr[nt].bf,
                    (short)0, accv[mt][nt], false, false);

#ifdef HAVE_ASYNC_LDS
        if (has_next) __builtin_amdgcn_s_wait_asynccnt(0);
#endif
        __syncthreads();
        buf ^= 1;
    }

    // ---- epilogue: bias (+ optional fused recon loss) ----
    float lsum = 0.0f;
#pragma unroll
    for (int nt = 0; nt < 4; ++nt) {
        int cn = n0 + (wn << 6) + (nt << 4) + rsel;
        if (cn < N) {
            float bv = bias[cn];
#pragma unroll
            for (int mt = 0; mt < 2; ++mt) {
                int cmb = m0 + (wm << 5) + (mt << 4) + ((lane >> 4) << 3);
#pragma unroll
                for (int r = 0; r < 8; ++r) {
                    float cv = accv[mt][nt][r] + bv;
                    if (MODE == 0) {
                        C[(size_t)(cmb + r) * N + cn] = cv;
                    } else {
                        float d = xin[(size_t)(cmb + r) * N + cn] - cv;
                        lsum += d * d + C_LOG2PI;
                    }
                }
            }
        }
    }
    if (MODE == 1) {
        red[tid] = lsum;
        __syncthreads();
        for (int s = 128; s > 0; s >>= 1) {
            if (tid < s) red[tid] += red[tid + s];
            __syncthreads();
        }
        if (tid == 0) atomicAdd(acc, -0.5f * red[0]);
    }
}

// ---------------------------------------------------------------------------
// Row-wise LayerNorm + SiLU, writing bf16 activation for the next GEMM.
// ---------------------------------------------------------------------------
__global__ __launch_bounds__(256)
void ln_silu_kernel(const float* __restrict__ C, const float* __restrict__ g,
                    const float* __restrict__ b2, ushort_t* __restrict__ out, int N) {
    __shared__ float red[256];
    const int row = blockIdx.x;
    const int tid = threadIdx.x;
    const float* cr = C + (size_t)row * N;

    float s = 0.0f;
    for (int j = tid; j < N; j += 256) s += cr[j];
    red[tid] = s; __syncthreads();
    for (int st = 128; st > 0; st >>= 1) { if (tid < st) red[tid] += red[tid + st]; __syncthreads(); }
    float mean = red[0] / (float)N;
    __syncthreads();

    float v = 0.0f;
    for (int j = tid; j < N; j += 256) { float d = cr[j] - mean; v += d * d; }
    red[tid] = v; __syncthreads();
    for (int st = 128; st > 0; st >>= 1) { if (tid < st) red[tid] += red[tid + st]; __syncthreads(); }
    float inv = rsqrtf(red[0] / (float)N + 1e-5f);

    for (int j = tid; j < N; j += 256) {
        float y = (cr[j] - mean) * inv * g[j] + b2[j];
        float sil = y / (1.0f + expf(-y));
        out[(size_t)row * N + j] = f2bf(sil);
    }
}

// ---------------------------------------------------------------------------
// Utility kernels
// ---------------------------------------------------------------------------
__global__ void cvt_kernel(const float* __restrict__ src, ushort_t* __restrict__ dst, int n) {
    for (int i = blockIdx.x * 256 + threadIdx.x; i < n; i += gridDim.x * 256)
        dst[i] = f2bf(src[i]);
}

__global__ void zero_kernel(float* __restrict__ p, int n) {
    for (int i = blockIdx.x * 256 + threadIdx.x; i < n; i += gridDim.x * 256)
        p[i] = 0.0f;
}

// Segment sums of z_ve / z_ev into per-var / per-env accumulators.
__global__ void seg1_kernel(const float* __restrict__ ve, const float* __restrict__ ev,
                            const int* __restrict__ vars, const int* __restrict__ envs,
                            float* sums_v, float* cnt_v, float* sums_e, float* cnt_e) {
    int t = blockIdx.x * 256 + threadIdx.x;
    if (t >= 8192 * 32) return;
    int i = t >> 5, j = t & 31;
    int v = vars[i], e = envs[i];
    atomicAdd(&sums_v[v * 32 + j], ve[(size_t)i * 64 + j]);
    atomicAdd(&sums_e[e * 32 + j], ev[(size_t)i * 64 + j]);
    if (j == 0) { atomicAdd(&cnt_v[v], 1.0f); atomicAdd(&cnt_e[e], 1.0f); }
}

__global__ void seg2v_kernel(const float* sums_v, const float* cnt_v,
                             const float* var_v_p, float* v_mu, float* acc) {
    int t = blockIdx.x * 256 + threadIdx.x;
    if (t >= 64 * 32) return;
    int v = t >> 5;
    float var_v = *var_v_p;
    float c = cnt_v[v];
    float mu = sums_v[t] / (c + var_v);
    v_mu[t] = mu;
    float lv = logf(var_v / (c + var_v));
    float term = -0.5f * (1.0f + lv - mu * mu - expf(lv)) * (1.0f / 32.0f) * (1.0f / 64.0f);
    atomicAdd(&acc[1], term);  // kld_v
}

__global__ void seg2e_kernel(const float* sums_e, const float* cnt_e,
                             const int* __restrict__ n_v_per_env,
                             const float* __restrict__ z_e_init,
                             const float* var_e_p, float* e_mu, float* acc) {
    int t = blockIdx.x * 256 + threadIdx.x;
    if (t >= 32 * 32) return;
    int e = t >> 5;
    float var_e = *var_e_p;
    float nv = (float)n_v_per_env[e];
    float c = cnt_e[e];
    float mu = sums_e[t] / (nv + var_e) + ((nv - c) / (nv + var_e)) * z_e_init[t];
    e_mu[t] = mu;
    float lv = logf(var_e / (nv + var_e));
    float term = -0.5f * (1.0f + lv - mu * mu - expf(lv)) * (1.0f / 32.0f) * (1.0f / 32.0f);
    atomicAdd(&acc[2], term);  // kld_e
}

__global__ __launch_bounds__(256)
void seg3_kernel(const float* __restrict__ ve, const float* __restrict__ ev,
                 const int* __restrict__ vars, const int* __restrict__ envs,
                 const float* __restrict__ v_mu, const float* __restrict__ e_mu,
                 const float* __restrict__ cnt_v, const float* __restrict__ cnt_e,
                 const float* var_v_p, const float* var_e_p, float* acc) {
    __shared__ float r1[256], r2[256];
    int t = blockIdx.x * 256 + threadIdx.x;
    int tid = threadIdx.x;
    float c_ve = 0.0f, c_ev = 0.0f;
    if (t < 8192 * 32) {
        int i = t >> 5, j = t & 31;
        int v = vars[i], e = envs[i];
        float var_v = *var_v_p, var_e = *var_e_p;

        float mu = ve[(size_t)i * 64 + j];
        float lv = ve[(size_t)i * 64 + 32 + j];
        float d  = mu - v_mu[v * 32 + j];
        float term = 1.0f - expf(lv) / var_v - (logf(var_v) - lv) - d * d / var_v;
        c_ve = (-0.5f / 32.0f) * term / (cnt_v[v] * 64.0f);

        float mue = ev[(size_t)i * 64 + j];
        float lve = ev[(size_t)i * 64 + 32 + j];
        float de  = mue - e_mu[e * 32 + j];
        float terme = 1.0f - expf(lve) / var_e - (logf(var_e) - lve) - de * de / var_e;
        c_ev = (-0.5f / 32.0f) * terme / (cnt_e[e] * 32.0f);
    }
    r1[tid] = c_ve; r2[tid] = c_ev; __syncthreads();
    for (int s = 128; s > 0; s >>= 1) {
        if (tid < s) { r1[tid] += r1[tid + s]; r2[tid] += r2[tid + s]; }
        __syncthreads();
    }
    if (tid == 0) { atomicAdd(&acc[3], r1[0]); atomicAdd(&acc[4], r2[0]); }
}

// z = concat(z_ve, z_ev) as bf16 decoder input
__global__ void build_z_kernel(const float* __restrict__ ve, const float* __restrict__ ev,
                               ushort_t* __restrict__ zbf) {
    int t = blockIdx.x * 256 + threadIdx.x;
    if (t >= 8192 * 64) return;
    int i = t >> 6, j = t & 63;
    float val = (j < 32) ? ve[(size_t)i * 64 + j] : ev[(size_t)i * 64 + (j - 32)];
    zbf[t] = f2bf(val);
}

__global__ void finalize_kernel(const float* __restrict__ acc, float* __restrict__ out) {
    if (threadIdx.x == 0 && blockIdx.x == 0) {
        out[0] = acc[0] / (8192.0f * 8192.0f);  // rec_term = mean(llkh)/B
        out[1] = acc[1];
        out[2] = acc[2];
        out[3] = acc[3];
        out[4] = acc[4];
    }
}

// ---------------------------------------------------------------------------
// Host orchestration
// ---------------------------------------------------------------------------
extern "C" void kernel_launch(void* const* d_in, const int* in_sizes, int n_in,
                              void* d_out, int out_size, void* d_ws, size_t ws_size,
                              hipStream_t stream) {
    (void)in_sizes; (void)n_in; (void)out_size; (void)ws_size;

    const float* x        = (const float*)d_in[0];
    const int*   vars     = (const int*)d_in[1];
    const int*   envs     = (const int*)d_in[2];
    const int*   n_v      = (const int*)d_in[3];
    const float* z_e_init = (const float*)d_in[4];
    const float* var_v_p  = (const float*)d_in[5];
    const float* var_e_p  = (const float*)d_in[6];
    const float* encW[4]  = {(const float*)d_in[7],  (const float*)d_in[8],
                             (const float*)d_in[9],  (const float*)d_in[10]};
    const float* encB[4]  = {(const float*)d_in[11], (const float*)d_in[12],
                             (const float*)d_in[13], (const float*)d_in[14]};
    const float* encG[4]  = {(const float*)d_in[15], (const float*)d_in[16],
                             (const float*)d_in[17], (const float*)d_in[18]};
    const float* encB2[4] = {(const float*)d_in[19], (const float*)d_in[20],
                             (const float*)d_in[21], (const float*)d_in[22]};
    const float* encLW = (const float*)d_in[23];
    const float* encLb = (const float*)d_in[24];
    const float* veW   = (const float*)d_in[25];
    const float* veb   = (const float*)d_in[26];
    const float* evW   = (const float*)d_in[27];
    const float* evb   = (const float*)d_in[28];
    const float* decW[4]  = {(const float*)d_in[29], (const float*)d_in[30],
                             (const float*)d_in[31], (const float*)d_in[32]};
    const float* decB[4]  = {(const float*)d_in[33], (const float*)d_in[34],
                             (const float*)d_in[35], (const float*)d_in[36]};
    const float* decG[4]  = {(const float*)d_in[37], (const float*)d_in[38],
                             (const float*)d_in[39], (const float*)d_in[40]};
    const float* decB2[4] = {(const float*)d_in[41], (const float*)d_in[42],
                             (const float*)d_in[43], (const float*)d_in[44]};
    const float* decLW = (const float*)d_in[45];
    const float* decLb = (const float*)d_in[46];

    float* out = (float*)d_out;

    // -------- workspace carve-out --------
    char* p = (char*)d_ws;
    auto alloc = [&](size_t bytes) -> void* {
        void* r = (void*)p;
        p += (bytes + 255) & ~(size_t)255;
        return r;
    };
    const size_t B = 8192;
    ushort_t* bufA     = (ushort_t*)alloc(B * 4096 * 2);  // ping (holds x_bf first)
    ushort_t* bufB     = (ushort_t*)alloc(B * 2048 * 2);  // pong
    float*    fbuf     = (float*)alloc(B * 2048 * 4);     // pre-LN GEMM out
    float*    out128   = (float*)alloc(B * 128 * 4);
    ushort_t* out128bf = (ushort_t*)alloc(B * 128 * 2);
    float*    vef      = (float*)alloc(B * 64 * 4);
    float*    evf      = (float*)alloc(B * 64 * 4);

    ushort_t* wb_e0   = (ushort_t*)alloc((size_t)2048 * 4096 * 2);
    ushort_t* wb_e1   = (ushort_t*)alloc((size_t)1024 * 2048 * 2);
    ushort_t* wb_e2   = (ushort_t*)alloc((size_t)512 * 1024 * 2);
    ushort_t* wb_e3   = (ushort_t*)alloc((size_t)256 * 512 * 2);
    ushort_t* wb_encL = (ushort_t*)alloc((size_t)128 * 256 * 2);
    ushort_t* wb_ve   = (ushort_t*)alloc((size_t)64 * 128 * 2);
    ushort_t* wb_ev   = (ushort_t*)alloc((size_t)64 * 128 * 2);
    ushort_t* wb_d0   = (ushort_t*)alloc((size_t)256 * 64 * 2);
    ushort_t* wb_d1   = (ushort_t*)alloc((size_t)512 * 256 * 2);
    ushort_t* wb_d2   = (ushort_t*)alloc((size_t)1024 * 512 * 2);
    ushort_t* wb_d3   = (ushort_t*)alloc((size_t)2048 * 1024 * 2);
    ushort_t* wb_decL = (ushort_t*)alloc((size_t)4096 * 2048 * 2);

    // stats block (contiguous, zeroed each launch)
    const int STATS_N = 2048 + 64 + 2048 + 1024 + 32 + 1024 + 5;
    float* stats  = (float*)alloc((size_t)STATS_N * 4);
    float* sums_v = stats;
    float* cnt_v  = stats + 2048;
    float* v_mu   = stats + 2048 + 64;
    float* sums_e = stats + 2048 + 64 + 2048;
    float* cnt_e  = stats + 2048 + 64 + 2048 + 1024;
    float* e_mu   = stats + 2048 + 64 + 2048 + 1024 + 32;
    float* acc    = stats + 2048 + 64 + 2048 + 1024 + 32 + 1024;

    auto cvt = [&](const float* s, ushort_t* d, size_t n) {
        int blocks = (int)((n + 255) / 256);
        if (blocks > 4096) blocks = 4096;
        cvt_kernel<<<blocks, 256, 0, stream>>>(s, d, (int)n);
    };
    auto gemm = [&](const ushort_t* A, const ushort_t* W, const float* bias,
                    float* C, int M, int N, int K) {
        dim3 grid((N + 127) / 128, M / 128);
        gemm_bf16_kernel<0><<<grid, 256, 0, stream>>>(A, W, bias, C, nullptr, nullptr, M, N, K);
    };
    auto ln = [&](const float* C, const float* g, const float* b2, ushort_t* o, int N) {
        ln_silu_kernel<<<8192, 256, 0, stream>>>(C, g, b2, o, N);
    };

    // -------- weight / input converts --------
    cvt(x, bufA, B * 4096);
    cvt(encW[0], wb_e0, (size_t)2048 * 4096);
    cvt(encW[1], wb_e1, (size_t)1024 * 2048);
    cvt(encW[2], wb_e2, (size_t)512 * 1024);
    cvt(encW[3], wb_e3, (size_t)256 * 512);
    cvt(encLW, wb_encL, (size_t)128 * 256);
    cvt(veW, wb_ve, (size_t)64 * 128);
    cvt(evW, wb_ev, (size_t)64 * 128);
    cvt(decW[0], wb_d0, (size_t)256 * 64);
    cvt(decW[1], wb_d1, (size_t)512 * 256);
    cvt(decW[2], wb_d2, (size_t)1024 * 512);
    cvt(decW[3], wb_d3, (size_t)2048 * 1024);
    cvt(decLW, wb_decL, (size_t)4096 * 2048);
    zero_kernel<<<(STATS_N + 255) / 256, 256, 0, stream>>>(stats, STATS_N);

    // -------- encoder --------
    gemm(bufA, wb_e0, encB[0], fbuf, 8192, 2048, 4096);
    ln(fbuf, encG[0], encB2[0], bufB, 2048);
    gemm(bufB, wb_e1, encB[1], fbuf, 8192, 1024, 2048);
    ln(fbuf, encG[1], encB2[1], bufA, 1024);
    gemm(bufA, wb_e2, encB[2], fbuf, 8192, 512, 1024);
    ln(fbuf, encG[2], encB2[2], bufB, 512);
    gemm(bufB, wb_e3, encB[3], fbuf, 8192, 256, 512);
    ln(fbuf, encG[3], encB2[3], bufA, 256);
    gemm(bufA, wb_encL, encLb, out128, 8192, 128, 256);
    cvt(out128, out128bf, B * 128);
    gemm(out128bf, wb_ve, veb, vef, 8192, 64, 128);
    gemm(out128bf, wb_ev, evb, evf, 8192, 64, 128);

    // -------- segment sums + KLDs --------
    seg1_kernel<<<(8192 * 32) / 256, 256, 0, stream>>>(vef, evf, vars, envs,
                                                       sums_v, cnt_v, sums_e, cnt_e);
    seg2v_kernel<<<8, 256, 0, stream>>>(sums_v, cnt_v, var_v_p, v_mu, acc);
    seg2e_kernel<<<4, 256, 0, stream>>>(sums_e, cnt_e, n_v, z_e_init, var_e_p, e_mu, acc);
    seg3_kernel<<<(8192 * 32) / 256, 256, 0, stream>>>(vef, evf, vars, envs, v_mu, e_mu,
                                                       cnt_v, cnt_e, var_v_p, var_e_p, acc);

    // -------- decoder --------
    build_z_kernel<<<(8192 * 64) / 256, 256, 0, stream>>>(vef, evf, bufA);  // z: 8192x64 bf16
    gemm(bufA, wb_d0, decB[0], fbuf, 8192, 256, 64);
    ln(fbuf, decG[0], decB2[0], bufB, 256);
    gemm(bufB, wb_d1, decB[1], fbuf, 8192, 512, 256);
    ln(fbuf, decG[1], decB2[1], bufA, 512);
    gemm(bufA, wb_d2, decB[2], fbuf, 8192, 1024, 512);
    ln(fbuf, decG[2], decB2[2], bufB, 1024);
    gemm(bufB, wb_d3, decB[3], fbuf, 8192, 2048, 1024);
    ln(fbuf, decG[3], decB2[3], bufA, 2048);
    // final layer with fused Gaussian recon loss (no recon materialization)
    {
        dim3 grid((4096 + 127) / 128, 8192 / 128);
        gemm_bf16_kernel<1><<<grid, 256, 0, stream>>>(bufA, wb_decL, decLb, nullptr,
                                                      x, &acc[0], 8192, 4096, 2048);
    }

    finalize_kernel<<<1, 32, 0, stream>>>(acc, out);
}
